// WindowAttention_52020643889406
// MI455X (gfx1250) — matrix operand order repaired and checked by reading, not compiled, not run on hardware
//
#include <hip/hip_runtime.h>
#include <hip/hip_bf16.h>
#include <stdint.h>

// ---------------- problem constants (from reference) ----------------
#define BB     4
#define CC     192
#define HH     224
#define WWD    224
#define NHH    6
#define HD     32
#define WSZ    8
#define TT     64            // tokens per window
#define SHF    4
#define WN     28            // 224/8 windows per dim
#define NWIN   (BB*WN*WN)    // 3136
#define CO     (3*CC)        // 576

// ---------------- WMMA fragment types (wave32, CDNA5) ----------------
typedef __attribute__((ext_vector_type(16))) __bf16 v16bf;
typedef __attribute__((ext_vector_type(8)))  __bf16 v8bf;
typedef __attribute__((ext_vector_type(8)))  float  v8f;

// native RTNE f32->bf16 (CDNA5 has hardware bf16 converts; clang lowers the
// fptrunc to v_cvt_*bf16_f32 instead of a 4-op integer rounding sequence)
__device__ __forceinline__ unsigned short f2bf(float f) {
    __bf16 h = (__bf16)f;
    return __builtin_bit_cast(unsigned short, h);
}

// build a v16bf A/B fragment from two 16-byte chunks
__device__ __forceinline__ v16bf ldfrag(const unsigned short* p0,
                                        const unsigned short* p1) {
    union { v16bf v; v8bf h[2]; } u;
    u.h[0] = *reinterpret_cast<const v8bf*>(p0);
    u.h[1] = *reinterpret_cast<const v8bf*>(p1);
    return u.v;
}

__device__ __forceinline__ v8f wmma_bf16(v16bf a, v16bf b, v8f c) {
    return __builtin_amdgcn_wmma_f32_16x16x32_bf16(
        false, a, false, b, (short)0, c, false, false);
}

// async LDS->global store, 16B per lane, tracked on ASYNCcnt
__device__ __forceinline__ void async_store_b128(uint32_t lds_off, float* gaddr) {
    asm volatile("global_store_async_from_lds_b128 %0, %1, off"
                 :: "v"((uint64_t)(uintptr_t)gaddr), "v"(lds_off)
                 : "memory");
}
__device__ __forceinline__ void wait_async0() {
    asm volatile("s_wait_asynccnt 0x0" ::: "memory");
}

// ---------------- prep: convert weights fp32 -> bf16 in d_ws ----------------
__global__ void cvt_weights(const float* __restrict__ wq, const float* __restrict__ wp,
                            unsigned short* __restrict__ wqb, unsigned short* __restrict__ wpb) {
    int i = blockIdx.x * 256 + threadIdx.x;
    if (i < CO * CC) wqb[i] = f2bf(wq[i]);
    if (i < CC * CC) wpb[i] = f2bf(wp[i]);
}

// ---------------- fused shifted-window attention: 1 block per window ----------------
// LDS layout (bytes), 156 KB total -> 2 workgroups per WGP:
//   Xs  [64][200] bf16      0      .. 25600   (window activations; reused as O bf16)
//   Qs  [64][200] bf16      25600  .. 51200   (pre-scaled by 1/sqrt(hd))
//   Ks  [64][200] bf16      51200  .. 76800
//   Vt  [192][72] bf16      76800  .. 104448  (V transposed: row c=head*32+d, col token)
//   Pb  [384][72] bf16      104448 .. 159744  (un-normalized exp; reused as Fs [192][68] f32)
#define OFF_Q  25600
#define OFF_K  51200
#define OFF_V  76800
#define OFF_P  104448
#define SMEM_BYTES 159744

__global__ void __launch_bounds__(256, 1)
swin_fused(const float* __restrict__ x,
           const unsigned short* __restrict__ wq,   // bf16 [576][192]
           const float* __restrict__ bq,
           const unsigned short* __restrict__ wp,   // bf16 [192][192]
           const float* __restrict__ bp,
           float* __restrict__ out) {
    extern __shared__ __align__(16) char smem[];
    unsigned short* Xs = (unsigned short*)(smem);
    unsigned short* Qs = (unsigned short*)(smem + OFF_Q);
    unsigned short* Ks = (unsigned short*)(smem + OFF_K);
    unsigned short* Vt = (unsigned short*)(smem + OFF_V);
    unsigned short* Pb = (unsigned short*)(smem + OFF_P);
    unsigned short* Os = Xs;                 // reuse after QKV phase
    float*          Fs = (float*)(smem + OFF_P);  // reuse after P*V: [192][68] f32

    const int wid = blockIdx.x;
    const int b   = wid / (WN * WN);
    const int wr  = wid % (WN * WN);
    const int wy  = wr / WN, wx = wr % WN;
    const int tid = threadIdx.x;
    const int lane = tid & 31, wave = tid >> 5;
    const int lo16 = lane & 15, hi = lane >> 4;     // hi in {0,1}

    // ---- phase 1: gather rolled window -> bf16 LDS (float4: shift keeps 4-pixel
    //      groups contiguous) ----
    for (int idx = tid; idx < CC * 8 * 2; idx += 256) {
        int g4 = idx & 1, ty = (idx >> 1) & 7, c = idx >> 4;
        int h  = wy * WSZ + ty + SHF; if (h >= HH)  h -= HH;
        int w0 = wx * WSZ + SHF + g4 * 4; if (w0 >= WWD) w0 -= WWD;
        const float4 v = *(const float4*)&x[((size_t)(b * CC + c) * HH + h) * WWD + w0];
        int t0 = ty * 8 + g4 * 4;
        Xs[(t0 + 0) * 200 + c] = f2bf(v.x);
        Xs[(t0 + 1) * 200 + c] = f2bf(v.y);
        Xs[(t0 + 2) * 200 + c] = f2bf(v.z);
        Xs[(t0 + 3) * 200 + c] = f2bf(v.w);
    }
    __syncthreads();

    // ---- phase 2: QKV GEMM, 2x2 register blocking (36 groups / 8 waves) ----
    // B fragments shared across both M tiles: halves weight refetch from L2.
    const float scale = 0.17677669529663687f;  // 1/sqrt(32)
    for (int tt = wave; tt < 36; tt += 8) {
        int mi2 = tt / 18, nj = tt % 18;       // nj uniform per wave
        int mbase = mi2 * 32, n0 = nj * 32;
        int na = n0 + lo16, nb = n0 + 16 + lo16;
        int mA0 = mbase + lo16, mA1 = mbase + 16 + lo16;
        v8f acc00 = {}, acc01 = {}, acc10 = {}, acc11 = {};
        #pragma unroll
        for (int kk = 0; kk < 6; ++kk) {
            int k0 = kk * 32;
            v16bf b0 = ldfrag(&wq[na * CC + k0 + hi * 16], &wq[na * CC + k0 + hi * 16 + 8]);
            v16bf b1 = ldfrag(&wq[nb * CC + k0 + hi * 16], &wq[nb * CC + k0 + hi * 16 + 8]);
            v16bf a0 = ldfrag(&Xs[mA0 * 200 + k0 + hi * 16], &Xs[mA0 * 200 + k0 + hi * 16 + 8]);
            v16bf a1 = ldfrag(&Xs[mA1 * 200 + k0 + hi * 16], &Xs[mA1 * 200 + k0 + hi * 16 + 8]);
            acc00 = wmma_bf16(a0, b0, acc00);
            acc01 = wmma_bf16(a0, b1, acc01);
            acc10 = wmma_bf16(a1, b0, acc10);
            acc11 = wmma_bf16(a1, b1, acc11);
        }
        float bias0 = bq[na], bias1 = bq[nb];
        if (nj < 6) {                               // Q, pre-scaled
            #pragma unroll
            for (int g = 0; g < 8; ++g) {
                int mr0 = mbase + g + hi * 8, mr1 = mr0 + 16;
                Qs[mr0 * 200 + na] = f2bf((acc00[g] + bias0) * scale);
                Qs[mr0 * 200 + nb] = f2bf((acc01[g] + bias1) * scale);
                Qs[mr1 * 200 + na] = f2bf((acc10[g] + bias0) * scale);
                Qs[mr1 * 200 + nb] = f2bf((acc11[g] + bias1) * scale);
            }
        } else if (nj < 12) {                       // K
            #pragma unroll
            for (int g = 0; g < 8; ++g) {
                int mr0 = mbase + g + hi * 8, mr1 = mr0 + 16;
                Ks[mr0 * 200 + (na - CC)] = f2bf(acc00[g] + bias0);
                Ks[mr0 * 200 + (nb - CC)] = f2bf(acc01[g] + bias1);
                Ks[mr1 * 200 + (na - CC)] = f2bf(acc10[g] + bias0);
                Ks[mr1 * 200 + (nb - CC)] = f2bf(acc11[g] + bias1);
            }
        } else {                                    // V, stored transposed
            int c0 = na - 2 * CC, c1 = nb - 2 * CC;
            #pragma unroll
            for (int g = 0; g < 8; ++g) {
                int mr0 = mbase + g + hi * 8, mr1 = mr0 + 16;
                Vt[c0 * 72 + mr0] = f2bf(acc00[g] + bias0);
                Vt[c1 * 72 + mr0] = f2bf(acc01[g] + bias1);
                Vt[c0 * 72 + mr1] = f2bf(acc10[g] + bias0);
                Vt[c1 * 72 + mr1] = f2bf(acc11[g] + bias1);
            }
        }
    }
    __syncthreads();

    // ---- phases 3-5 merged: scores + in-register softmax + P*V ----
    // One wave owns a full 16-row x 64-col score block, so softmax reduces
    // across the 16-lane half with shfl_xor; no f32 score buffer, no barrier.
    for (int tt = wave; tt < 24; tt += 8) {
        int hh = tt >> 2, mi = tt & 3;
        int m0 = mi * 16, m = m0 + lo16;

        // scores: 4 N-tiles sharing one A-fragment (K = hd = 32 -> 1 wmma each)
        v16bf aq = ldfrag(&Qs[m * 200 + hh * 32 + hi * 16],
                          &Qs[m * 200 + hh * 32 + hi * 16 + 8]);
        v8f s0 = {}, s1 = {}, s2 = {}, s3 = {};
        {
            v16bf k0f = ldfrag(&Ks[(0*16 + lo16) * 200 + hh * 32 + hi * 16],
                               &Ks[(0*16 + lo16) * 200 + hh * 32 + hi * 16 + 8]);
            v16bf k1f = ldfrag(&Ks[(1*16 + lo16) * 200 + hh * 32 + hi * 16],
                               &Ks[(1*16 + lo16) * 200 + hh * 32 + hi * 16 + 8]);
            v16bf k2f = ldfrag(&Ks[(2*16 + lo16) * 200 + hh * 32 + hi * 16],
                               &Ks[(2*16 + lo16) * 200 + hh * 32 + hi * 16 + 8]);
            v16bf k3f = ldfrag(&Ks[(3*16 + lo16) * 200 + hh * 32 + hi * 16],
                               &Ks[(3*16 + lo16) * 200 + hh * 32 + hi * 16 + 8]);
            s0 = wmma_bf16(aq, k0f, s0);
            s1 = wmma_bf16(aq, k1f, s1);
            s2 = wmma_bf16(aq, k2f, s2);
            s3 = wmma_bf16(aq, k3f, s3);
        }

        // in-register row softmax; keep 1/sum per lane (g-th value is the row
        // this lane stores in the P*V epilogue: row = m0 + g + 8*hi)
        float invs[8];
        #pragma unroll
        for (int g = 0; g < 8; ++g) {
            float mx = fmaxf(fmaxf(s0[g], s1[g]), fmaxf(s2[g], s3[g]));
            mx = fmaxf(mx, __shfl_xor(mx, 1));
            mx = fmaxf(mx, __shfl_xor(mx, 2));
            mx = fmaxf(mx, __shfl_xor(mx, 4));
            mx = fmaxf(mx, __shfl_xor(mx, 8));
            float e0 = __expf(s0[g] - mx), e1 = __expf(s1[g] - mx);
            float e2 = __expf(s2[g] - mx), e3 = __expf(s3[g] - mx);
            float sum = e0 + e1 + e2 + e3;
            sum += __shfl_xor(sum, 1);
            sum += __shfl_xor(sum, 2);
            sum += __shfl_xor(sum, 4);
            sum += __shfl_xor(sum, 8);
            invs[g] = 1.0f / sum;
            int row = hh * 64 + m0 + g + hi * 8;
            Pb[row * 72 +      lo16] = f2bf(e0);
            Pb[row * 72 + 16 + lo16] = f2bf(e1);
            Pb[row * 72 + 32 + lo16] = f2bf(e2);
            Pb[row * 72 + 48 + lo16] = f2bf(e3);
        }

        // P*V for the same row group (same-wave LDS RAW is in-order)
        v8f o0 = {}, o1 = {};
        #pragma unroll
        for (int k0 = 0; k0 < 64; k0 += 32) {
            v16bf ap = ldfrag(&Pb[(hh * 64 + m) * 72 + k0 + hi * 16],
                              &Pb[(hh * 64 + m) * 72 + k0 + hi * 16 + 8]);
            v16bf vb0 = ldfrag(&Vt[(hh * 32 + lo16) * 72 + k0 + hi * 16],
                               &Vt[(hh * 32 + lo16) * 72 + k0 + hi * 16 + 8]);
            v16bf vb1 = ldfrag(&Vt[(hh * 32 + 16 + lo16) * 72 + k0 + hi * 16],
                               &Vt[(hh * 32 + 16 + lo16) * 72 + k0 + hi * 16 + 8]);
            o0 = wmma_bf16(ap, vb0, o0);
            o1 = wmma_bf16(ap, vb1, o1);
        }
        #pragma unroll
        for (int g = 0; g < 8; ++g) {
            int mr = m0 + g + hi * 8;
            Os[mr * 200 + hh * 32 + lo16]      = f2bf(o0[g] * invs[g]);
            Os[mr * 200 + hh * 32 + 16 + lo16] = f2bf(o1[g] * invs[g]);
        }
    }
    __syncthreads();

    // ---- phase 6: proj GEMM, 2 N-tiles per A-fragment -> Fs [channel][token] f32 ----
    for (int tt = wave; tt < 24; tt += 8) {
        int mi = tt / 6, nj = tt % 6;
        int m0 = mi * 16, n0 = nj * 32;
        int m = m0 + lo16;
        int na = n0 + lo16, nb = n0 + 16 + lo16;
        v8f acc0 = {}, acc1 = {};
        #pragma unroll
        for (int kk = 0; kk < 6; ++kk) {
            int k0 = kk * 32;
            v16bf a  = ldfrag(&Os[m * 200 + k0 + hi * 16],
                              &Os[m * 200 + k0 + hi * 16 + 8]);
            v16bf b0 = ldfrag(&wp[na * CC + k0 + hi * 16],
                              &wp[na * CC + k0 + hi * 16 + 8]);
            v16bf b1 = ldfrag(&wp[nb * CC + k0 + hi * 16],
                              &wp[nb * CC + k0 + hi * 16 + 8]);
            acc0 = wmma_bf16(a, b0, acc0);
            acc1 = wmma_bf16(a, b1, acc1);
        }
        float bias0 = bp[na], bias1 = bp[nb];
        // D tile: VGPR g holds row m0+g+8*hi -> 8 consecutive tokens -> 2x float4
        float* d0 = &Fs[na * 68 + m0 + hi * 8];
        float* d1 = &Fs[nb * 68 + m0 + hi * 8];
        *(float4*)(d0)     = make_float4(acc0[0]+bias0, acc0[1]+bias0, acc0[2]+bias0, acc0[3]+bias0);
        *(float4*)(d0 + 4) = make_float4(acc0[4]+bias0, acc0[5]+bias0, acc0[6]+bias0, acc0[7]+bias0);
        *(float4*)(d1)     = make_float4(acc1[0]+bias1, acc1[1]+bias1, acc1[2]+bias1, acc1[3]+bias1);
        *(float4*)(d1 + 4) = make_float4(acc1[4]+bias1, acc1[5]+bias1, acc1[6]+bias1, acc1[7]+bias1);
    }
    __syncthreads();

    // ---- phase 7: async LDS->global scatter with +SHIFT un-roll ----
    // Each op moves 4 consecutive pixels (16B) of one (channel,row); the +4
    // shift keeps every 4-pixel group contiguous in global memory.
    for (int idx = tid; idx < CC * 8 * 2; idx += 256) {
        int g4 = idx & 1;
        int ty = (idx >> 1) & 7;
        int c  = idx >> 4;
        int h  = wy * WSZ + ty + SHF; if (h >= HH) h -= HH;
        int w0 = wx * WSZ + SHF + g4 * 4; if (w0 >= WWD) w0 -= WWD;
        uint32_t lds = (uint32_t)(uintptr_t)&Fs[c * 68 + ty * 8 + g4 * 4];
        float* gp = &out[((size_t)(b * CC + c) * HH + h) * WWD + w0];
        async_store_b128(lds, gp);
    }
    wait_async0();   // explicit drain (s_endpgm would also wait-idle)
}

extern "C" void kernel_launch(void* const* d_in, const int* in_sizes, int n_in,
                              void* d_out, int out_size, void* d_ws, size_t ws_size,
                              hipStream_t stream) {
    const float* x     = (const float*)d_in[0];
    const float* wqkv  = (const float*)d_in[1];
    const float* bqkv  = (const float*)d_in[2];
    const float* wproj = (const float*)d_in[3];
    const float* bproj = (const float*)d_in[4];
    float* out = (float*)d_out;

    unsigned short* wqb = (unsigned short*)d_ws;          // 576*192 bf16
    unsigned short* wpb = wqb + CO * CC;                  // 192*192 bf16

    cvt_weights<<<(CO * CC + 255) / 256, 256, 0, stream>>>(wqkv, wproj, wqb, wpb);
    swin_fused<<<NWIN, 256, SMEM_BYTES, stream>>>(x, wqb, bqkv, wpb, bproj, out);
}
